// CustomLSTM_1614907703621
// MI455X (gfx1250) — matrix operand order repaired and checked by reading, not compiled, not run on hardware
//
#include <hip/hip_runtime.h>
#include <stdint.h>

// ---------------------------------------------------------------------------
// 2-layer LSTM for MI455X (gfx1250, wave32, WMMA).
//   B=128, T=256, F=1024, H=1024, gates=4H=4096.
// GEMMs in bf16 with f32 accumulation via v_wmma_f32_16x16x32_bf16.
// Weight tiles staged to LDS with GLOBAL_LOAD_ASYNC_TO_LDS_B128 (ASYNCcnt),
// double-buffered, one barrier per 32-K chunk.
// One fused cell kernel per (layer, timestep); h ping-pong buffered.
// ---------------------------------------------------------------------------

typedef __bf16 bf16;
typedef __attribute__((ext_vector_type(8)))  __bf16 v8bf;
typedef __attribute__((ext_vector_type(16))) __bf16 v16bf;
typedef __attribute__((ext_vector_type(8)))  float  v8f;

constexpr int kB = 128;
constexpr int kT = 256;
constexpr int kF = 1024;
constexpr int kH = 1024;
constexpr int kG = 4 * kH;   // 4096 gate outputs

__device__ __forceinline__ float fast_sigmoid(float x) {
    return 1.0f / (1.0f + __expf(-x));
}
__device__ __forceinline__ float fast_tanh(float x) {
    float ax = fabsf(x);
    float e  = __expf(-2.0f * ax);
    float t  = (1.0f - e) / (1.0f + e);
    return copysignf(t, x);
}

__device__ __forceinline__ v16bf combine16(v8bf lo, v8bf hi) {
    v16bf r;
#pragma unroll
    for (int i = 0; i < 8; ++i) { r[i] = lo[i]; r[i + 8] = hi[i]; }
    return r;
}

// Stage one 64-row x 32-K bf16 weight tile (4 gates x 16 n-cols) into LDS
// using async copies: 128 threads x 32 bytes. ASYNCcnt-tracked; the 24-bit
// instruction offset applies to BOTH the global and LDS addresses.
__device__ __forceinline__ void stage_async(const bf16* __restrict__ W, int n0,
                                            int kc, bf16* dstBase, int tid) {
    const int r  = tid >> 1;          // 0..63 : gate*16 + n
    const int qe = (tid & 1) * 16;    // element offset 0 / 16 within the row
    const bf16* src = W + (size_t)((r >> 4) * kH + n0 + (r & 15)) * 1024 + kc + qe;
    uint32_t lds = (uint32_t)(uintptr_t)(dstBase + r * 32 + qe);
    uint64_t g   = (uint64_t)(uintptr_t)src;
    asm volatile("global_load_async_to_lds_b128 %0, %1, off"
                 :: "v"(lds), "v"(g) : "memory");
    asm volatile("global_load_async_to_lds_b128 %0, %1, off offset:16"
                 :: "v"(lds), "v"(g) : "memory");
}

__device__ __forceinline__ void wait_async_barrier() {
    asm volatile("s_wait_asynccnt 0x0" ::: "memory");
    __syncthreads();
}

// ---------------------------------------------------------------------------
// Fused LSTM cell step.
//   gates(B x 4H) = [xin | hprev] (B x 2048) @ [Win | Whh]^T + bias
//   then i,f,g,o activations, c/h update, optional fp32 y output.
//
// Grid:  (H/16)   Block: 128 threads = 4 waves (wave32).
// Wave w computes a 32(M) x 16(N) tile for EACH of the 4 gates:
//   rows  m0 = w*32 .. +32  (batch; 4 waves cover all 128 rows)
//   cols  n0 = blockIdx.x*16 .. +16 (hidden index)
//   gate g uses weight rows g*H + n0 .. +16.
// B-fragments are shared by all 4 waves -> async-staged via LDS, 2 buffers.
// ---------------------------------------------------------------------------
__global__ __launch_bounds__(128) void lstm_cell_kernel(
    const bf16* __restrict__ xin,  int ldx,         // (B x 1024), row stride ldx
    const bf16* __restrict__ hprev,                  // (B x H) bf16
    const bf16* __restrict__ Win,                    // (4H x 1024) bf16 row-major
    const bf16* __restrict__ Whh,                    // (4H x H) bf16 row-major
    const float* __restrict__ bias,                  // (4H) = b_ih + b_hh
    float* __restrict__ c,                           // (B x H) f32, in/out
    bf16* __restrict__ hout,                         // (B x H) bf16 (next step in)
    float* __restrict__ yout, int ldy)               // optional (B x T x H) slice
{
    __shared__ bf16 sB[2][64 * 32];   // double-buffered weight tile (4 KB each)

    const int tid  = threadIdx.x;
    const int lane = tid & 31;
    const int wave = tid >> 5;
    const int n0   = blockIdx.x * 16;   // hidden-column tile
    const int m0   = wave * 32;         // batch-row tile (2 x 16 M sub-tiles)

    const int nlane = lane & 15;   // N column / M row within tile
    const int half  = lane >> 4;   // lane half selects K (A/B) or M (C/D) group

    // --- accumulators = bias broadcast (bias depends only on gate column) ---
    v8f acc[2][4];
#pragma unroll
    for (int g = 0; g < 4; ++g) {
        float bv = bias[g * kH + n0 + nlane];
#pragma unroll
        for (int mt = 0; mt < 2; ++mt)
#pragma unroll
            for (int r = 0; r < 8; ++r) acc[mt][g][r] = bv;
    }

    // --- prologue: stage K-chunk 0 ---
    stage_async(Win, n0, 0, &sB[0][0], tid);
    wait_async_barrier();

    // --- flattened K loop: chunks 0..31 = input GEMM, 32..63 = recurrent ---
    for (int kidx = 0; kidx < 64; ++kidx) {
        const int p = kidx & 1;

        // prefetch next chunk into the other buffer (overlaps with WMMAs)
        if (kidx + 1 < 64) {
            const bf16* Wn = (kidx + 1 < 32) ? Win : Whh;
            stage_async(Wn, n0, ((kidx + 1) & 31) * 32, &sB[p ^ 1][0], tid);
        }

        // A fragments: two 16x32 bf16 tiles (rows m0.. and m0+16..)
        const bf16* A  = (kidx < 32) ? xin : hprev;
        const int   ld = (kidx < 32) ? ldx : kH;
        const int   kc = (kidx & 31) * 32;
        v16bf aF[2];
#pragma unroll
        for (int mt = 0; mt < 2; ++mt) {
            const bf16* arow = A + (size_t)(m0 + mt * 16 + nlane) * ld + kc;
            v8bf alo = *(const v8bf*)(arow + half * 16);
            v8bf ahi = *(const v8bf*)(arow + half * 16 + 8);
            aF[mt] = combine16(alo, ahi);
        }

        // 4 gates x 2 M-tiles = 8 WMMAs per chunk; B frag reused across M.
#pragma unroll
        for (int g = 0; g < 4; ++g) {
            v16bf bF =
                *(const v16bf*)(&sB[p][0] + (g * 16 + nlane) * 32 + half * 16);
#pragma unroll
            for (int mt = 0; mt < 2; ++mt) {
                acc[mt][g] = __builtin_amdgcn_wmma_f32_16x16x32_bf16(
                    false, aF[mt], false, bF, (short)0, acc[mt][g], false, false);
            }
        }

        wait_async_barrier();
    }

    // --- epilogue: activations + c/h update, per-lane elementwise ---
    // C/D layout: lane holds col n, rows mbase..mbase+7 in VGPRs 0..7.
    const int n = n0 + nlane;
#pragma unroll
    for (int mt = 0; mt < 2; ++mt) {
        const int mbase = m0 + mt * 16 + 8 * half;
#pragma unroll
        for (int r = 0; r < 8; ++r) {
            const int m = mbase + r;
            float iv = fast_sigmoid(acc[mt][0][r]);
            float fv = fast_sigmoid(acc[mt][1][r]);
            float gv = fast_tanh(acc[mt][2][r]);
            float ov = fast_sigmoid(acc[mt][3][r]);
            size_t ci = (size_t)m * kH + n;
            float cn = fv * c[ci] + iv * gv;
            c[ci] = cn;
            float hn = ov * fast_tanh(cn);
            hout[ci] = (bf16)hn;
            if (yout) yout[(size_t)m * ldy + n] = hn;
        }
    }
}

// ---------------------------------------------------------------------------
// Helper kernels
// ---------------------------------------------------------------------------
__global__ void f32_to_bf16_kernel(const float* __restrict__ src,
                                   bf16* __restrict__ dst, size_t n) {
    size_t i = (size_t)blockIdx.x * blockDim.x + threadIdx.x;
    size_t stride = (size_t)gridDim.x * blockDim.x;
    for (; i < n; i += stride) dst[i] = (bf16)src[i];
}

__global__ void bias_sum_kernel(const float* __restrict__ a,
                                const float* __restrict__ b,
                                float* __restrict__ o, int n) {
    int i = blockIdx.x * blockDim.x + threadIdx.x;
    if (i < n) o[i] = a[i] + b[i];
}

__global__ void zero_u32_kernel(uint32_t* __restrict__ p, size_t n) {
    size_t i = (size_t)blockIdx.x * blockDim.x + threadIdx.x;
    size_t stride = (size_t)gridDim.x * blockDim.x;
    for (; i < n; i += stride) p[i] = 0u;
}

// ---------------------------------------------------------------------------
// Host launcher
// ---------------------------------------------------------------------------
extern "C" void kernel_launch(void* const* d_in, const int* in_sizes, int n_in,
                              void* d_out, int out_size, void* d_ws, size_t ws_size,
                              hipStream_t stream) {
    const float* x    = (const float*)d_in[0];
    const float* Wih0 = (const float*)d_in[1];
    const float* bih0 = (const float*)d_in[2];
    const float* Whh0 = (const float*)d_in[3];
    const float* bhh0 = (const float*)d_in[4];
    const float* Wih1 = (const float*)d_in[5];
    const float* bih1 = (const float*)d_in[6];
    const float* Whh1 = (const float*)d_in[7];
    const float* bhh1 = (const float*)d_in[8];
    float* y = (float*)d_out;

    // --- workspace layout ---
    char* ws = (char*)d_ws;
    size_t off = 0;
    auto alloc = [&](size_t bytes) -> char* {
        char* p = ws + off;
        off += (bytes + 255) & ~(size_t)255;
        return p;
    };
    bf16* xb  = (bf16*)alloc((size_t)kB * kT * kF * 2);   // 64 MB
    bf16* w0i = (bf16*)alloc((size_t)kG * kF * 2);        // 8 MB each
    bf16* w0h = (bf16*)alloc((size_t)kG * kH * 2);
    bf16* w1i = (bf16*)alloc((size_t)kG * kH * 2);
    bf16* w1h = (bf16*)alloc((size_t)kG * kH * 2);
    float* bias0 = (float*)alloc((size_t)kG * 4);
    float* bias1 = (float*)alloc((size_t)kG * 4);
    // state region (zeroed every call): h0[2], h1[2] bf16 + c0, c1 f32
    char* state = ws + off;
    bf16* h0[2]; bf16* h1[2];
    h0[0] = (bf16*)alloc((size_t)kB * kH * 2);
    h0[1] = (bf16*)alloc((size_t)kB * kH * 2);
    h1[0] = (bf16*)alloc((size_t)kB * kH * 2);
    h1[1] = (bf16*)alloc((size_t)kB * kH * 2);
    float* c0 = (float*)alloc((size_t)kB * kH * 4);
    float* c1 = (float*)alloc((size_t)kB * kH * 4);
    size_t state_bytes = (char*)(c1 + (size_t)kB * kH) - state;

    // --- one-time (per call) conversions: fp32 -> bf16, bias sums, zero state
    f32_to_bf16_kernel<<<2048, 256, 0, stream>>>(x,    xb,  (size_t)kB * kT * kF);
    f32_to_bf16_kernel<<<1024, 256, 0, stream>>>(Wih0, w0i, (size_t)kG * kF);
    f32_to_bf16_kernel<<<1024, 256, 0, stream>>>(Whh0, w0h, (size_t)kG * kH);
    f32_to_bf16_kernel<<<1024, 256, 0, stream>>>(Wih1, w1i, (size_t)kG * kH);
    f32_to_bf16_kernel<<<1024, 256, 0, stream>>>(Whh1, w1h, (size_t)kG * kH);
    bias_sum_kernel<<<(kG + 255) / 256, 256, 0, stream>>>(bih0, bhh0, bias0, kG);
    bias_sum_kernel<<<(kG + 255) / 256, 256, 0, stream>>>(bih1, bhh1, bias1, kG);
    zero_u32_kernel<<<512, 256, 0, stream>>>((uint32_t*)state, state_bytes / 4);

    // --- time recurrence: 2 fused cell kernels per step, h ping-pong ---
    const dim3 grid(kH / 16);
    for (int t = 0; t < kT; ++t) {
        const int cur = t & 1, nxt = cur ^ 1;
        // layer 0: input = x_t (row stride T*F), h0 double-buffered
        lstm_cell_kernel<<<grid, 128, 0, stream>>>(
            xb + (size_t)t * kF, kT * kF, h0[cur],
            w0i, w0h, bias0, c0, h0[nxt], nullptr, 0);
        // layer 1: input = fresh h0, writes y_t (row stride T*H)
        lstm_cell_kernel<<<grid, 128, 0, stream>>>(
            h0[nxt], kH, h1[cur],
            w1i, w1h, bias1, c1, h1[nxt],
            y + (size_t)t * kH, kT * kH);
    }
}
